// Decoder_32452772888819
// MI455X (gfx1250) — matrix operand order repaired and checked by reading, not compiled, not run on hardware
//
#include <hip/hip_runtime.h>
#include <hip/hip_bf16.h>
#include <cstdint>

// ---------------------------------------------------------------------------
// Problem constants (match reference)
// ---------------------------------------------------------------------------
#define BATCH   64
#define NPIX    196          // 14*14
#define HDIM    512
#define ADIM    512
#define EDIM    512
#define VOCAB   10000
#define MAXCAP  50
#define TSTEPS  49           // MAXCAP - 1

typedef __attribute__((ext_vector_type(16))) __bf16 v16bf;
typedef __attribute__((ext_vector_type(8)))  float  v8f;

static __device__ __forceinline__ __bf16 f2bf(float f) { return (__bf16)f; }

// ---------------------------------------------------------------------------
// bf16-WMMA GEMM:  C[M,N] (+)= A[M,K] * W[N,K]^T + bias[N]
// One wave per (MT*16)x16 output strip: the B fragment is loaded once per
// K-step and reused across MT WMMAs (MT accumulators), which is optimal for
// skinny-M GEMMs where the N*K weight matrix dominates traffic.
// grid = (N/16, M/(MT*16)).  Requires M%(16*MT)==0, N%16==0, K%32==0.
// ---------------------------------------------------------------------------
template<int MT>
__global__ __launch_bounds__(32)
void wmma_gemm_kernel(const float* __restrict__ A, int lda,
                      const float* __restrict__ W, int ldw,
                      const float* __restrict__ bias,
                      float* __restrict__ C, int ldc,
                      int K, int accum)
{
    const int lane = threadIdx.x & 31;
    const int n0   = blockIdx.x << 4;
    const int m0   = blockIdx.y * (MT << 4);
    const int loc  = lane & 15;        // A: row-in-tile, B: col-in-tile
    const int hi   = lane >> 4;        // lane half
    const int kha  = hi << 3;          // A K-offset for upper lane half (+8)
    const int khb  = hi << 4;          // unused: B also uses kha (+8), same K order as A

    v8f acc[MT];
#pragma unroll
    for (int mt = 0; mt < MT; ++mt) acc[mt] = (v8f){};
    if (accum) {
#pragma unroll
        for (int mt = 0; mt < MT; ++mt)
#pragma unroll
            for (int r = 0; r < 8; ++r)
                acc[mt][r] = C[(size_t)(m0 + (mt << 4) + (hi << 3) + r) * ldc + n0 + loc];
    }

    const float* __restrict__ Arow[MT];
#pragma unroll
    for (int mt = 0; mt < MT; ++mt)
        Arow[mt] = A + (size_t)(m0 + (mt << 4) + loc) * lda;
    const float* __restrict__ Wrow = W + (size_t)(n0 + loc) * ldw;

    for (int k0 = 0; k0 < K; k0 += 32) {
        // ---- B fragment (32x16 bf16): same K order as A: v<4 -> K=2v (+kha); v>=4 -> K=16+2(v-4) (+kha)
        v16bf bv;
#pragma unroll
        for (int v = 0; v < 8; ++v) {
            const float2 w2 = *(const float2*)(Wrow + k0 + kha + ((v < 4) ? (v << 1) : (16 + ((v - 4) << 1))));
            bv[2 * v]     = f2bf(w2.x);
            bv[2 * v + 1] = f2bf(w2.y);
        }
        // prefetch next K-block of the weight stream while WMMAs execute
        if (k0 + 256 < K)
            __builtin_prefetch(Wrow + k0 + 256, 0, 1);

        // ---- MT A fragments, each fused with the shared B fragment
#pragma unroll
        for (int mt = 0; mt < MT; ++mt) {
            v16bf av;
#pragma unroll
            for (int v = 0; v < 8; ++v) {
                // A 16x32 bf16 layout: v<4 -> K=2v (+kha); v>=4 -> K=16+2(v-4) (+kha)
                const int ka = k0 + kha + ((v < 4) ? (v << 1) : (16 + ((v - 4) << 1)));
                const float2 a2 = *(const float2*)(Arow[mt] + ka);
                av[2 * v]     = f2bf(a2.x);
                av[2 * v + 1] = f2bf(a2.y);
            }
            acc[mt] = __builtin_amdgcn_wmma_f32_16x16x32_bf16(
                          /*neg_a=*/false, av, /*neg_b=*/false, bv,
                          /*c_mod=*/(short)0, acc[mt], /*reuse_a=*/false, /*reuse_b=*/false);
        }
    }

    const float bb = bias ? bias[n0 + loc] : 0.0f;
#pragma unroll
    for (int mt = 0; mt < MT; ++mt)
#pragma unroll
        for (int r = 0; r < 8; ++r)
            C[(size_t)(m0 + (mt << 4) + (hi << 3) + r) * ldc + n0 + loc] = acc[mt][r] + bb;
}

// ---------------------------------------------------------------------------
// Prep: stable descending sort of caplens (argsort(-cl)), gather caps_s,
// emit int outputs (as float) to tail of d_out.
// ---------------------------------------------------------------------------
__global__ __launch_bounds__(64)
void prep_kernel(const int* __restrict__ caps, const int* __restrict__ caplens,
                 int* __restrict__ sort_ind, int* __restrict__ dec_len,
                 int* __restrict__ caps_s,
                 float* __restrict__ out_caps, float* __restrict__ out_declen,
                 float* __restrict__ out_sortind)
{
    __shared__ int si[BATCH];
    const int tid = threadIdx.x;
    if (tid == 0) {
        int pos = 0;
        for (int key = MAXCAP; key >= 0 && pos < BATCH; --key)
            for (int b = 0; b < BATCH; ++b)
                if (caplens[b] == key) si[pos++] = b;
    }
    __syncthreads();
    if (tid < BATCH) {
        const int s = si[tid];
        sort_ind[tid]   = s;
        const int dl    = caplens[s] - 1;
        dec_len[tid]    = dl;
        out_declen[tid] = (float)dl;
        out_sortind[tid] = (float)s;
    }
    __syncthreads();
    for (int i = tid; i < BATCH * MAXCAP; i += 64) {
        const int b = i / MAXCAP, j = i % MAXCAP;
        const int v = caps[si[b] * MAXCAP + j];
        caps_s[i]   = v;
        out_caps[i] = (float)v;
    }
}

__global__ __launch_bounds__(128)
void gather_feats_kernel(const float* __restrict__ imf, const int* __restrict__ sort_ind,
                         float* __restrict__ feats)
{
    const int p = blockIdx.x, b = blockIdx.y;
    const float* in  = imf   + ((size_t)sort_ind[b] * NPIX + p) * HDIM;
    float*       out = feats + ((size_t)b * NPIX + p) * HDIM;
    for (int h = threadIdx.x; h < HDIM; h += 128) out[h] = in[h];
}

__global__ __launch_bounds__(128)
void gather_emb_kernel(const float* __restrict__ emb_table, const int* __restrict__ caps_s,
                       float* __restrict__ emb)
{
    const int t = blockIdx.x, b = blockIdx.y;
    const int tok = caps_s[b * MAXCAP + t];
    const float* in  = emb_table + (size_t)tok * EDIM;
    float*       out = emb + ((size_t)b * TSTEPS + t) * EDIM;
    for (int h = threadIdx.x; h < EDIM; h += 128) out[h] = in[h];
}

__global__ __launch_bounds__(256)
void zero_hc_kernel(float* __restrict__ h, float* __restrict__ c)
{
    const int i = blockIdx.x * 256 + threadIdx.x;
    if (i < BATCH * HDIM) { h[i] = 0.f; c[i] = 0.f; }
}

// ---------------------------------------------------------------------------
// Attention: e = relu(enc_att + att2) . W_full ; alpha = softmax_p(e);
// ctx = alpha @ feats ; x = [emb_t, ctx]. One block per batch row.
// ---------------------------------------------------------------------------
__global__ __launch_bounds__(256)
void attention_kernel(const float* __restrict__ enc_att,
                      const float* __restrict__ att2,
                      const float* __restrict__ feats,
                      const float* __restrict__ emb,
                      const float* __restrict__ W_full,
                      const float* __restrict__ b_full,
                      float* __restrict__ x, float* __restrict__ alpha_out, int t)
{
    __shared__ float att2_s[ADIM];
    __shared__ float e_s[NPIX];
    __shared__ float red[256];

    const int b = blockIdx.x;
    const int tid = threadIdx.x;
    const int lane = tid & 31, wave = tid >> 5;

    for (int a = tid; a < ADIM; a += 256) att2_s[a] = att2[b * ADIM + a];
    __syncthreads();

    const float* ea = enc_att + (size_t)b * NPIX * ADIM;
    const float bf0 = b_full[0];
    for (int p = wave; p < NPIX; p += 8) {
        const float* row = ea + (size_t)p * ADIM;
        float s = 0.f;
        for (int a = lane; a < ADIM; a += 32)
            s += fmaxf(row[a] + att2_s[a], 0.f) * W_full[a];
#pragma unroll
        for (int off = 16; off > 0; off >>= 1) s += __shfl_xor(s, off, 32);
        if (lane == 0) e_s[p] = s + bf0;
    }
    __syncthreads();

    // softmax over 196 positions
    float m = -INFINITY;
    for (int p = tid; p < NPIX; p += 256) m = fmaxf(m, e_s[p]);
    red[tid] = m; __syncthreads();
    for (int s2 = 128; s2 > 0; s2 >>= 1) {
        if (tid < s2) red[tid] = fmaxf(red[tid], red[tid + s2]);
        __syncthreads();
    }
    const float mx = red[0]; __syncthreads();

    float ssum = 0.f;
    for (int p = tid; p < NPIX; p += 256) {
        const float ev = __expf(e_s[p] - mx);
        e_s[p] = ev; ssum += ev;
    }
    red[tid] = ssum; __syncthreads();
    for (int s2 = 128; s2 > 0; s2 >>= 1) {
        if (tid < s2) red[tid] += red[tid + s2];
        __syncthreads();
    }
    const float inv = 1.f / red[0]; __syncthreads();

    for (int p = tid; p < NPIX; p += 256) {
        e_s[p] *= inv;
        alpha_out[b * NPIX + p] = e_s[p];
    }
    __syncthreads();

    // ctx = alpha @ feats ; pack x = [emb_t, ctx]
    const float* fb = feats + (size_t)b * NPIX * HDIM;
    for (int h = tid; h < HDIM; h += 256) {
        float s = 0.f;
        for (int p = 0; p < NPIX; ++p) s += e_s[p] * fb[(size_t)p * HDIM + h];
        x[(size_t)b * (EDIM + HDIM) + EDIM + h] = s;
        x[(size_t)b * (EDIM + HDIM) + h] = emb[((size_t)b * TSTEPS + t) * EDIM + h];
    }
}

// ---------------------------------------------------------------------------
// LSTM cell elementwise + masked state update
// ---------------------------------------------------------------------------
__global__ __launch_bounds__(256)
void lstm_cell_kernel(const float* __restrict__ gates,
                      float* __restrict__ h, float* __restrict__ c,
                      float* __restrict__ h_new, float* __restrict__ c_new,
                      const int* __restrict__ dec_len, int t)
{
    const int b = blockIdx.x;
    const bool upd = (t < dec_len[b]);
    const float* g = gates + (size_t)b * 4 * HDIM;
    for (int j = threadIdx.x; j < HDIM; j += 256) {
        const float gi = g[j];
        const float gf = g[HDIM + j];
        const float gg = g[2 * HDIM + j];
        const float go = g[3 * HDIM + j];
        const float si = 1.f / (1.f + __expf(-gi));
        const float sf = 1.f / (1.f + __expf(-gf));
        const float so = 1.f / (1.f + __expf(-go));
        const float cn = sf * c[(size_t)b * HDIM + j] + si * tanhf(gg);
        const float hn = so * tanhf(cn);
        c_new[(size_t)b * HDIM + j] = cn;
        h_new[(size_t)b * HDIM + j] = hn;
        if (upd) { c[(size_t)b * HDIM + j] = cn; h[(size_t)b * HDIM + j] = hn; }
    }
}

// ---------------------------------------------------------------------------
// Masked store of step outputs (zeros where mask is false, matching reference)
// ---------------------------------------------------------------------------
__global__ __launch_bounds__(256)
void store_step_kernel(const float* __restrict__ score, const float* __restrict__ alpha,
                       const int* __restrict__ dec_len,
                       float* __restrict__ out_scores, float* __restrict__ out_weights, int t)
{
    const int b = blockIdx.x;
    const bool mvalid = (t < dec_len[b]);
    const int idx = blockIdx.y * 256 + threadIdx.x;
    if (idx < VOCAB)
        out_scores[((size_t)b * TSTEPS + t) * VOCAB + idx] =
            mvalid ? score[(size_t)b * VOCAB + idx] : 0.f;
    if (blockIdx.y == 0 && idx < NPIX)
        out_weights[((size_t)b * TSTEPS + t) * NPIX + idx] =
            mvalid ? alpha[b * NPIX + idx] : 0.f;
}

// ---------------------------------------------------------------------------
// Host-side launcher
// ---------------------------------------------------------------------------
extern "C" void kernel_launch(void* const* d_in, const int* in_sizes, int n_in,
                              void* d_out, int out_size, void* d_ws, size_t ws_size,
                              hipStream_t stream)
{
    const float* image_features = (const float*)d_in[0];
    const int*   caps           = (const int*)  d_in[1];
    const int*   caplens        = (const int*)  d_in[2];
    const float* emb_table      = (const float*)d_in[3];
    const float* W_ih           = (const float*)d_in[4];
    const float* W_hh           = (const float*)d_in[5];
    const float* b_ih           = (const float*)d_in[6];
    const float* b_hh           = (const float*)d_in[7];
    const float* W_enc          = (const float*)d_in[8];
    const float* b_enc          = (const float*)d_in[9];
    const float* W_dec          = (const float*)d_in[10];
    const float* b_dec          = (const float*)d_in[11];
    const float* W_full         = (const float*)d_in[12];
    const float* b_full         = (const float*)d_in[13];
    const float* W_score        = (const float*)d_in[14];
    const float* b_score        = (const float*)d_in[15];

    // ---- workspace carve-up -------------------------------------------------
    char* ws = (char*)d_ws;
    int* sort_ind = (int*)ws;  ws += BATCH * sizeof(int);
    int* dec_len  = (int*)ws;  ws += BATCH * sizeof(int);
    int* caps_s   = (int*)ws;  ws += BATCH * MAXCAP * sizeof(int);
    float* feats   = (float*)ws; ws += (size_t)BATCH * NPIX * HDIM * sizeof(float);
    float* enc_att = (float*)ws; ws += (size_t)BATCH * NPIX * ADIM * sizeof(float);
    float* embb    = (float*)ws; ws += (size_t)BATCH * TSTEPS * EDIM * sizeof(float);
    float* att2    = (float*)ws; ws += (size_t)BATCH * ADIM * sizeof(float);
    float* xbuf    = (float*)ws; ws += (size_t)BATCH * (EDIM + HDIM) * sizeof(float);
    float* alpha   = (float*)ws; ws += (size_t)BATCH * NPIX * sizeof(float);
    float* gates   = (float*)ws; ws += (size_t)BATCH * 4 * HDIM * sizeof(float);
    float* hbuf    = (float*)ws; ws += (size_t)BATCH * HDIM * sizeof(float);
    float* cbuf    = (float*)ws; ws += (size_t)BATCH * HDIM * sizeof(float);
    float* h_new   = (float*)ws; ws += (size_t)BATCH * HDIM * sizeof(float);
    float* c_new   = (float*)ws; ws += (size_t)BATCH * HDIM * sizeof(float);
    float* score   = (float*)ws; ws += (size_t)BATCH * VOCAB * sizeof(float);

    // ---- output carve-up (scores, caps_s, decode_lengths, weights, sort_ind)
    float* out         = (float*)d_out;
    float* out_scores  = out;
    float* out_caps    = out_scores + (size_t)BATCH * TSTEPS * VOCAB;
    float* out_declen  = out_caps + (size_t)BATCH * MAXCAP;
    float* out_weights = out_declen + BATCH;
    float* out_sortind = out_weights + (size_t)BATCH * TSTEPS * NPIX;

    // ---- one-time prep ------------------------------------------------------
    prep_kernel<<<1, 64, 0, stream>>>(caps, caplens, sort_ind, dec_len, caps_s,
                                      out_caps, out_declen, out_sortind);
    gather_feats_kernel<<<dim3(NPIX, BATCH), 128, 0, stream>>>(image_features, sort_ind, feats);
    gather_emb_kernel<<<dim3(TSTEPS, BATCH), 128, 0, stream>>>(emb_table, caps_s, embb);
    zero_hc_kernel<<<(BATCH * HDIM + 255) / 256, 256, 0, stream>>>(hbuf, cbuf);

    // enc_att[b,p,:] = feats @ W_enc^T + b_enc   (M = 64*196 = 12544)
    wmma_gemm_kernel<4><<<dim3(ADIM / 16, (BATCH * NPIX) / 64), 32, 0, stream>>>(
        feats, HDIM, W_enc, HDIM, b_enc, enc_att, ADIM, HDIM, 0);

    // ---- time-step loop -----------------------------------------------------
    for (int t = 0; t < TSTEPS; ++t) {
        // att2 = h @ W_dec^T + b_dec   (64 x 512)
        wmma_gemm_kernel<4><<<dim3(ADIM / 16, 1), 32, 0, stream>>>(
            hbuf, HDIM, W_dec, HDIM, b_dec, att2, ADIM, HDIM, 0);

        attention_kernel<<<BATCH, 256, 0, stream>>>(
            enc_att, att2, feats, embb, W_full, b_full, xbuf, alpha, t);

        // gates = x @ W_ih^T + b_ih   (64 x 2048, K = 1024)
        wmma_gemm_kernel<4><<<dim3((4 * HDIM) / 16, 1), 32, 0, stream>>>(
            xbuf, EDIM + HDIM, W_ih, EDIM + HDIM, b_ih, gates, 4 * HDIM, EDIM + HDIM, 0);
        // gates += h @ W_hh^T + b_hh  (K = 512)
        wmma_gemm_kernel<4><<<dim3((4 * HDIM) / 16, 1), 32, 0, stream>>>(
            hbuf, HDIM, W_hh, HDIM, b_hh, gates, 4 * HDIM, HDIM, 1);

        lstm_cell_kernel<<<BATCH, 256, 0, stream>>>(gates, hbuf, cbuf, h_new, c_new, dec_len, t);

        // score = h_new @ W_score^T + b_score   (64 x 10000, 625 column strips)
        wmma_gemm_kernel<4><<<dim3(VOCAB / 16, 1), 32, 0, stream>>>(
            h_new, HDIM, W_score, HDIM, b_score, score, VOCAB, HDIM, 0);

        store_step_kernel<<<dim3(BATCH, (VOCAB + 255) / 256), 256, 0, stream>>>(
            score, alpha, dec_len, out_scores, out_weights, t);
    }
}
